// BLSTM_CRF1_57853209477467
// MI455X (gfx1250) — matrix operand order repaired and checked by reading, not compiled, not run on hardware
//
#include <hip/hip_runtime.h>
#include <hip/hip_bf16.h>
#include <math.h>

// ---------------------------------------------------------------------------
// BiLSTM-CRF for MI455X (gfx1250, wave32).
// Heavy GEMM -> v_wmma_f32_16x16x32_bf16 with 4x2 register blocking.
// Recurrence -> persistent WG + LDS broadcast.  Viterbi -> single wave.
// ---------------------------------------------------------------------------

#define L_DIM 4096
#define E_DIM 1024
#define HH_DIM 512
#define G_DIM 2048        // 4*HH
#define N_CAT 4096        // both directions' gates
#define T_TAGS 21

typedef __attribute__((ext_vector_type(16))) __bf16 v16bf_t;
typedef __attribute__((ext_vector_type(8)))  float  v8f_t;

// ---------------- conversion / gather kernels ------------------------------

__global__ void __launch_bounds__(256)
gather_embed_bf16(const int* __restrict__ sent, const float* __restrict__ emb,
                  __bf16* __restrict__ xbf)
{
    int idx = blockIdx.x * 256 + threadIdx.x;      // over L*E
    int l = idx >> 10;                             // /E_DIM
    int e = idx & (E_DIM - 1);
    xbf[idx] = (__bf16)emb[sent[l] * E_DIM + e];
}

__global__ void __launch_bounds__(256)
f32_to_bf16(const float* __restrict__ src, __bf16* __restrict__ dst, int n)
{
    int idx = blockIdx.x * 256 + threadIdx.x;
    if (idx < n) dst[idx] = (__bf16)src[idx];
}

__global__ void __launch_bounds__(256)
build_bias_cat(const float* __restrict__ bf, const float* __restrict__ bb,
               float* __restrict__ bias)
{
    int i = blockIdx.x * 256 + threadIdx.x;        // 0..4095
    bias[i] = (i < G_DIM) ? bf[i] : bb[i - G_DIM];
}

// ---------------- input-projection GEMM via WMMA ---------------------------
// xp[L][4096] = xbf[L][1024] @ Wcat^T + bias_cat
// Block = 256 threads = 8 waves tiling a 128x128 macro-tile.
// Each wave: 64x32 output (4 M-tiles x 2 N-tiles) -> 8 WMMA per K-chunk,
// 12 b128 loads per K-chunk (1.5 loads/WMMA).

__global__ void __launch_bounds__(256)
input_gemm_wmma(const __bf16* __restrict__ X,     // L x E
                const __bf16* __restrict__ W,     // N_CAT x E (row-major)
                const float*  __restrict__ bias,  // N_CAT
                float* __restrict__ out)          // L x N_CAT
{
    const int tid    = threadIdx.x;
    const int lane   = tid & 31;
    const int wave   = tid >> 5;                  // 0..7
    const int wave_m = wave >> 2;                 // 0..1  (64 rows each)
    const int wave_n = wave & 3;                  // 0..3  (32 cols each)
    const int half   = lane >> 4;                 // 0/1
    const int lm     = lane & 15;

    const int m0 = blockIdx.x * 128 + wave_m * 64;   // wave's first output row
    const int n0 = blockIdx.y * 128 + wave_n * 32;   // wave's first output col

    const __bf16* Arow[4];
    const __bf16* Brow[2];
    #pragma unroll
    for (int mi = 0; mi < 4; ++mi)
        Arow[mi] = X + (size_t)(m0 + mi * 16 + lm) * E_DIM;
    #pragma unroll
    for (int ni = 0; ni < 2; ++ni)
        Brow[ni] = W + (size_t)(n0 + ni * 16 + lm) * E_DIM;

    v8f_t acc[4][2];
    #pragma unroll
    for (int mi = 0; mi < 4; ++mi)
        #pragma unroll
        for (int ni = 0; ni < 2; ++ni)
            acc[mi][ni] = (v8f_t){};

    union Frag { v16bf_t v; unsigned int u[8]; };

    for (int k0 = 0; k0 < E_DIM; k0 += 32) {
        Frag a[4], b[2];
        // A-fragment: 16-bit A 16x32 layout (ISA 7.12.2)
        #pragma unroll
        for (int mi = 0; mi < 4; ++mi)
            #pragma unroll
            for (int i = 0; i < 8; ++i) {
                int k = k0 + ((i >> 2) << 4) + (half << 3) + ((i & 3) << 1);
                a[mi].u[i] = *(const unsigned int*)(Arow[mi] + k);
            }
        // B-fragment: 16-bit B 32x16 layout, same K order as A (lanes 0-15 K=0..7,16..23; 16-31 K=8..15,24..31)
        #pragma unroll
        for (int ni = 0; ni < 2; ++ni)
            #pragma unroll
            for (int j = 0; j < 8; ++j) {
                int k = k0 + ((j >> 2) << 4) + (half << 3) + ((j & 3) << 1);
                b[ni].u[j] = *(const unsigned int*)(Brow[ni] + k);
            }
        #pragma unroll
        for (int mi = 0; mi < 4; ++mi)
            #pragma unroll
            for (int ni = 0; ni < 2; ++ni)
                acc[mi][ni] = __builtin_amdgcn_wmma_f32_16x16x32_bf16(
                    /*neg_a=*/false, a[mi].v, /*neg_b=*/false, b[ni].v,
                    /*c_mod=*/(short)0, acc[mi][ni],
                    /*reuse_a=*/false, /*reuse_b=*/false);
    }

    // C/D layout: lane n = lane%16; VGPR r -> M = r + 8*half
    #pragma unroll
    for (int ni = 0; ni < 2; ++ni) {
        const int n  = n0 + ni * 16 + lm;
        const float bn = bias[n];
        #pragma unroll
        for (int mi = 0; mi < 4; ++mi) {
            #pragma unroll
            for (int r = 0; r < 8; ++r) {
                int m = m0 + mi * 16 + r + (half << 3);
                out[(size_t)m * N_CAT + n] = acc[mi][ni][r] + bn;
            }
        }
    }
}

// ---------------- sequential LSTM (one WG per direction) -------------------

__device__ __forceinline__ float sigmoidf_(float x) {
    return 1.0f / (1.0f + __expf(-x));
}

__global__ void __launch_bounds__(1024)
lstm_seq(const float* __restrict__ xp,          // L x 4096
         const __bf16* __restrict__ whh_all,    // 2 x 2048 x 512 bf16
         const float* __restrict__ b_hh_f,
         const float* __restrict__ b_hh_b,
         const float* __restrict__ h0,          // 2 x 512
         const float* __restrict__ c0,          // 2 x 512
         float* __restrict__ lstm_out)          // L x 1024
{
    __shared__ float h_s[HH_DIM];
    __shared__ float c_s[HH_DIM];
    __shared__ float g_s[G_DIM];

    const int dir = blockIdx.x;                 // 0 fwd, 1 bwd
    const int tid = threadIdx.x;                // 0..1023
    const float* b_hh = dir ? b_hh_b : b_hh_f;
    const __bf16* whh = whh_all + (size_t)dir * G_DIM * HH_DIM;

    if (tid < HH_DIM) {
        h_s[tid] = h0[dir * HH_DIM + tid];
        c_s[tid] = c0[dir * HH_DIM + tid];
    }
    __syncthreads();

    const __bf16* w0 = whh + (size_t)tid * HH_DIM;
    const __bf16* w1 = whh + (size_t)(tid + 1024) * HH_DIM;
    const float bh0 = b_hh[tid];
    const float bh1 = b_hh[tid + 1024];

    for (int s = 0; s < L_DIM; ++s) {
        const int t = dir ? (L_DIM - 1 - s) : s;
        const float* xpt = xp + (size_t)t * N_CAT + dir * G_DIM;

        float acc0 = xpt[tid] + bh0;
        float acc1 = xpt[tid + 1024] + bh1;

        __builtin_prefetch(w0, 0, 1);           // -> global_prefetch_b8
        __builtin_prefetch(w1, 0, 1);

        #pragma unroll 8
        for (int k = 0; k < HH_DIM; ++k) {
            float hv = h_s[k];                  // LDS broadcast across lanes
            acc0 = fmaf((float)w0[k], hv, acc0);
            acc1 = fmaf((float)w1[k], hv, acc1);
        }
        g_s[tid]        = acc0;
        g_s[tid + 1024] = acc1;
        __syncthreads();

        if (tid < HH_DIM) {
            float ig = sigmoidf_(g_s[tid]);
            float fg = sigmoidf_(g_s[HH_DIM + tid]);
            float gg = tanhf(g_s[2 * HH_DIM + tid]);
            float og = sigmoidf_(g_s[3 * HH_DIM + tid]);
            float c  = fg * c_s[tid] + ig * gg;
            float h  = og * tanhf(c);
            c_s[tid] = c;
            h_s[tid] = h;
            lstm_out[(size_t)t * (2 * HH_DIM) + dir * HH_DIM + tid] = h;
        }
        __syncthreads();
    }
}

// ---------------- emission GEMV --------------------------------------------

__global__ void __launch_bounds__(32)
emission_kernel(const float* __restrict__ lstm_out,  // L x 1024
                const float* __restrict__ W_out,     // 21 x 1024
                const float* __restrict__ b_out,     // 21
                float* __restrict__ emission)        // L x 21
{
    const int t   = blockIdx.x;
    const int tag = threadIdx.x;
    if (tag < T_TAGS) {
        const float* h = lstm_out + (size_t)t * (2 * HH_DIM);
        const float* w = W_out + (size_t)tag * (2 * HH_DIM);
        float acc = b_out[tag];
        #pragma unroll 8
        for (int k = 0; k < 2 * HH_DIM; ++k) acc = fmaf(h[k], w[k], acc);
        emission[t * T_TAGS + tag] = acc;
    }
}

// ---------------- Viterbi (single wave) ------------------------------------

__global__ void __launch_bounds__(32)
viterbi_kernel(const float* __restrict__ emission,   // L x 21
               const float* __restrict__ transition, // 21 x 22
               int* __restrict__ bptr,               // L x 21 (ws)
               float* __restrict__ out)              // [score, path(L)]
{
    __shared__ float fv[T_TAGS];
    __shared__ float trans[T_TAGS * T_TAGS];

    const int j = threadIdx.x;
    if (j < T_TAGS) {
        for (int i = 0; i < T_TAGS; ++i)
            trans[j * T_TAGS + i] = transition[j * (T_TAGS + 1) + i];
        fv[j] = transition[j * (T_TAGS + 1) + T_TAGS] + emission[j];
    }
    __syncthreads();

    for (int t = 1; t < L_DIM; ++t) {
        float best = -3.4e38f;
        int   bi   = 0;
        if (j < T_TAGS) {
            #pragma unroll
            for (int i = 0; i < T_TAGS; ++i) {
                float sc = fv[i] + trans[j * T_TAGS + i];
                if (sc > best) { best = sc; bi = i; }
            }
            bptr[t * T_TAGS + j] = bi;
        }
        __syncthreads();
        if (j < T_TAGS) fv[j] = best + emission[t * T_TAGS + j];
        __syncthreads();
    }

    if (j == 0) {
        int   best = 0;
        float bs   = fv[0];
        for (int i = 1; i < T_TAGS; ++i)
            if (fv[i] > bs) { bs = fv[i]; best = i; }
        out[0]             = bs;
        out[1 + L_DIM - 1] = (float)best;
        int tag = best;
        for (int t = L_DIM - 1; t >= 1; --t) {
            tag = bptr[t * T_TAGS + tag];
            out[1 + t - 1] = (float)tag;
        }
    }
}

// ---------------------------------------------------------------------------

extern "C" void kernel_launch(void* const* d_in, const int* in_sizes, int n_in,
                              void* d_out, int out_size, void* d_ws, size_t ws_size,
                              hipStream_t stream)
{
    const int*   sentence   = (const int*)  d_in[0];
    const float* emb        = (const float*)d_in[1];
    const float* W_ih_f     = (const float*)d_in[2];
    const float* W_hh_f     = (const float*)d_in[3];
    const float* b_ih_f     = (const float*)d_in[4];
    const float* b_hh_f     = (const float*)d_in[5];
    const float* W_ih_b     = (const float*)d_in[6];
    const float* W_hh_b     = (const float*)d_in[7];
    const float* b_ih_b     = (const float*)d_in[8];
    const float* b_hh_b     = (const float*)d_in[9];
    const float* W_out      = (const float*)d_in[10];
    const float* b_out      = (const float*)d_in[11];
    const float* transition = (const float*)d_in[12];
    const float* h0         = (const float*)d_in[13];
    const float* c0         = (const float*)d_in[14];
    float* out = (float*)d_out;

    // ---- workspace carve-up (256B aligned) ----
    char* ws = (char*)d_ws;
    size_t off = 0;
    auto carve = [&](size_t bytes) {
        char* p = ws + off;
        off += (bytes + 255) & ~(size_t)255;
        return p;
    };
    __bf16* xbf      = (__bf16*)carve((size_t)L_DIM * E_DIM * 2);          // 8 MB
    __bf16* wcat_bf  = (__bf16*)carve((size_t)N_CAT * E_DIM * 2);          // 8 MB
    __bf16* whh_bf   = (__bf16*)carve((size_t)2 * G_DIM * HH_DIM * 2);     // 4 MB
    float*  bias_cat = (float*) carve((size_t)N_CAT * 4);                  // 16 KB
    float*  xp       = (float*) carve((size_t)L_DIM * N_CAT * 4);          // 64 MB
    float*  lstm_o   = (float*) carve((size_t)L_DIM * 2 * HH_DIM * 4);     // 16 MB
    float*  emis     = (float*) carve((size_t)L_DIM * T_TAGS * 4);         // 344 KB
    int*    bptr     = (int*)   carve((size_t)L_DIM * T_TAGS * 4);         // 344 KB

    // 1) gather + fp32->bf16 conversions
    gather_embed_bf16<<<(L_DIM * E_DIM) / 256, 256, 0, stream>>>(sentence, emb, xbf);
    f32_to_bf16<<<(G_DIM * E_DIM) / 256, 256, 0, stream>>>(W_ih_f, wcat_bf, G_DIM * E_DIM);
    f32_to_bf16<<<(G_DIM * E_DIM) / 256, 256, 0, stream>>>(W_ih_b, wcat_bf + (size_t)G_DIM * E_DIM, G_DIM * E_DIM);
    f32_to_bf16<<<(G_DIM * HH_DIM) / 256, 256, 0, stream>>>(W_hh_f, whh_bf, G_DIM * HH_DIM);
    f32_to_bf16<<<(G_DIM * HH_DIM) / 256, 256, 0, stream>>>(W_hh_b, whh_bf + (size_t)G_DIM * HH_DIM, G_DIM * HH_DIM);
    build_bias_cat<<<N_CAT / 256, 256, 0, stream>>>(b_ih_f, b_ih_b, bias_cat);

    // 2) big input-projection GEMM on WMMA (34 GFLOP), 128x128 macro-tiles
    dim3 gemm_grid(L_DIM / 128, N_CAT / 128);
    input_gemm_wmma<<<gemm_grid, 256, 0, stream>>>(xbf, wcat_bf, bias_cat, xp);

    // 3) sequential bidirectional LSTM: one persistent WG per direction
    lstm_seq<<<2, 1024, 0, stream>>>(xp, whh_bf, b_hh_f, b_hh_b, h0, c0, lstm_o);

    // 4) emission scores
    emission_kernel<<<L_DIM, 32, 0, stream>>>(lstm_o, W_out, b_out, emis);

    // 5) Viterbi decode (single wave) + backtrack into d_out
    viterbi_kernel<<<1, 32, 0, stream>>>(emis, transition, bptr, out);
}
